// MultiHeadAttention_59863254172512
// MI455X (gfx1250) — matrix operand order repaired and checked by reading, not compiled, not run on hardware
//
#include <hip/hip_runtime.h>

typedef __attribute__((ext_vector_type(16))) __bf16 v16bf;
typedef __attribute__((ext_vector_type(8)))  float  v8f;

union Frag16 { uint4 u[2]; unsigned us[8]; v16bf v; };
union Pack8  { __bf16 h[8]; uint4 u; };

__device__ inline v8f wmma_bf16(v16bf a, v16bf b, v8f c) {
    return __builtin_amdgcn_wmma_f32_16x16x32_bf16(false, a, false, b, (short)0, c, false, false);
}

__device__ inline unsigned pk_bf16(float x, float y) {
    union { __bf16 h[2]; unsigned u; } t;
    t.h[0] = (__bf16)x; t.h[1] = (__bf16)y;
    return t.u;
}

// ---------------------------------------------------------------------------
// fp32 -> bf16 elementwise cast (vectorized x4)
// ---------------------------------------------------------------------------
__global__ __launch_bounds__(256) void cast_bf16_kernel(const float* __restrict__ in,
                                                        __bf16* __restrict__ out, int n4) {
    int i = blockIdx.x * blockDim.x + threadIdx.x;
    if (i < n4) {
        float4 f = ((const float4*)in)[i];
        union { __bf16 h[4]; uint2 u; } t;
        t.h[0] = (__bf16)f.x; t.h[1] = (__bf16)f.y;
        t.h[2] = (__bf16)f.z; t.h[3] = (__bf16)f.w;
        ((uint2*)out)[i] = t.u;
    }
}

// ---------------------------------------------------------------------------
// fp32 [R,C] -> bf16 [C,R] tiled transpose (LDS)
// ---------------------------------------------------------------------------
__global__ __launch_bounds__(256) void transpose_cast_kernel(const float* __restrict__ in,
                                                             __bf16* __restrict__ outT,
                                                             int R, int C) {
    __shared__ float tile[32][33];
    int bx = blockIdx.x * 32, by = blockIdx.y * 32;
    int tx = threadIdx.x, ty = threadIdx.y;  // 32 x 8
    #pragma unroll
    for (int i = 0; i < 32; i += 8)
        tile[ty + i][tx] = in[(size_t)(by + ty + i) * C + bx + tx];
    __syncthreads();
    #pragma unroll
    for (int i = 0; i < 32; i += 8)
        outT[(size_t)(bx + ty + i) * R + by + tx] = (__bf16)tile[tx][ty + i];
}

// ---------------------------------------------------------------------------
// Generic bf16 GEMM: C = A[M,K] @ Bt[N,K]^T,  accum fp32, out bf16.
// Block: 256 thr (8 waves), tile 64(M) x 256(N); wave tile 32x64.
// mode 0: C row-major [M,N]
// mode 1: attention-V transposed store: C[((b*16+h)*64+dv)*Lk + lk]
// ---------------------------------------------------------------------------
__global__ __launch_bounds__(256) void gemm_bf16_kernel(const __bf16* __restrict__ A,
                                                        const __bf16* __restrict__ Bt,
                                                        __bf16* __restrict__ C,
                                                        int M, int N, int K,
                                                        int mode, int Lk) {
    const int wave = threadIdx.x >> 5;
    const int lane = threadIdx.x & 31;
    const int l15 = lane & 15, lh = lane >> 4;
    const int wm = wave >> 2, wn = wave & 3;
    const int m0 = blockIdx.y * 64 + wm * 32;
    const int n0 = blockIdx.x * 256 + wn * 64;

    v8f acc[2][4];
    const v8f zero = {0.f, 0.f, 0.f, 0.f, 0.f, 0.f, 0.f, 0.f};
    #pragma unroll
    for (int mi = 0; mi < 2; mi++)
        #pragma unroll
        for (int ni = 0; ni < 4; ni++) acc[mi][ni] = zero;

    for (int k0 = 0; k0 < K; k0 += 32) {
        Frag16 a[2], b[4];
        #pragma unroll
        for (int mi = 0; mi < 2; mi++) {
            const __bf16* ap = A + (size_t)(m0 + mi * 16 + l15) * K + k0 + 8 * lh;
            a[mi].u[0] = *(const uint4*)ap;
            a[mi].u[1] = *(const uint4*)(ap + 16);
        }
        #pragma unroll
        for (int ni = 0; ni < 4; ni++) {
            const __bf16* bp = Bt + (size_t)(n0 + ni * 16 + l15) * K + k0 + 8 * lh;
            b[ni].u[0] = *(const uint4*)bp;
            b[ni].u[1] = *(const uint4*)(bp + 16);
        }
        #pragma unroll
        for (int mi = 0; mi < 2; mi++)
            #pragma unroll
            for (int ni = 0; ni < 4; ni++)
                acc[mi][ni] = wmma_bf16(a[mi].v, b[ni].v, acc[mi][ni]);
    }

    if (mode == 0) {
        #pragma unroll
        for (int mi = 0; mi < 2; mi++)
            #pragma unroll
            for (int ni = 0; ni < 4; ni++)
                #pragma unroll
                for (int r = 0; r < 8; r++) {
                    int row = m0 + mi * 16 + r + 8 * lh;
                    int col = n0 + ni * 16 + l15;
                    C[(size_t)row * N + col] = (__bf16)acc[mi][ni][r];
                }
    } else {
        #pragma unroll
        for (int mi = 0; mi < 2; mi++) {
            int mrow0 = m0 + mi * 16 + 8 * lh;
            int bidx = mrow0 / Lk, lk0 = mrow0 % Lk;
            #pragma unroll
            for (int ni = 0; ni < 4; ni++) {
                int n = n0 + ni * 16 + l15;
                int hh = n >> 6, dv = n & 63;
                Pack8 p;
                #pragma unroll
                for (int r = 0; r < 8; r++) p.h[r] = (__bf16)acc[mi][ni][r];
                __bf16* outp = C + ((size_t)(bidx * 16 + hh) * 64 + dv) * Lk + lk0;
                *(uint4*)outp = p.u;
            }
        }
    }
}

// ---------------------------------------------------------------------------
// Flash attention, transposed form. One wave = 16 queries x 1 head.
// S^T = K @ Q^T (keys on accumulator rows -> in-lane softmax reductions),
// O^T = V^T @ P^T with P re-packed as B operand (in-lane pack, no shuffle).
// Two value streams share one attention matrix.
// ---------------------------------------------------------------------------
__global__ __launch_bounds__(128) void attn_kernel(const __bf16* __restrict__ Qb,
                                                   const __bf16* __restrict__ Kb,
                                                   const __bf16* __restrict__ VtS,
                                                   const __bf16* __restrict__ VtO,
                                                   const float* __restrict__ scores,
                                                   const int* __restrict__ mask,
                                                   __bf16* __restrict__ Os,
                                                   __bf16* __restrict__ Oo) {
    const int gw = blockIdx.x * 4 + (threadIdx.x >> 5);
    const int qt = gw & 127, h = (gw >> 7) & 15, b = gw >> 11;
    const int lane = threadIdx.x & 31;
    const int l15 = lane & 15, lh = lane >> 4;
    const int q0 = qt * 16;

    // Q as B-operand fragments: lane = query l15, K-dim chunk = d (8 at 8*lh, 8 at 16+8*lh)
    Frag16 qf[2];
    const __bf16* qbase = Qb + (size_t)(b * 2048 + q0 + l15) * 1024 + h * 64;
    #pragma unroll
    for (int d = 0; d < 2; d++) {
        const __bf16* p = qbase + d * 32 + 8 * lh;
        qf[d].u[0] = *(const uint4*)p;
        qf[d].u[1] = *(const uint4*)(p + 16);
    }

    const v8f zero = {0.f, 0.f, 0.f, 0.f, 0.f, 0.f, 0.f, 0.f};
    v8f oS[4], oO[4];
    #pragma unroll
    for (int i = 0; i < 4; i++) { oS[i] = zero; oO[i] = zero; }
    float mrun = -3.0e38f, lrun = 0.0f;

    const float* scb = scores + b * 2048;
    const int*   mkb = mask + (size_t)(b * 2048 + q0 + l15) * 2048;

    for (int k0 = 0; k0 < 2048; k0 += 32) {
        // --- S^T tiles (16 keys each), K-dim = head dim 64 ---
        v8f s[2] = {zero, zero};
        #pragma unroll
        for (int t = 0; t < 2; t++) {
            const __bf16* kp = Kb + (size_t)(b * 2048 + k0 + t * 16 + l15) * 1024 + h * 64;
            #pragma unroll
            for (int d = 0; d < 2; d++) {
                Frag16 kf;
                const __bf16* p = kp + d * 32 + 8 * lh;
                kf.u[0] = *(const uint4*)p;
                kf.u[1] = *(const uint4*)(p + 16);
                s[t] = wmma_bf16(kf.v, qf[d].v, s[t]);
            }
        }
        // --- bias, mask, online softmax ---
        float xv[2][8];
        float tmax = -3.0e38f;
        #pragma unroll
        for (int t = 0; t < 2; t++) {
            int ko = k0 + t * 16 + 8 * lh;
            float4 sc0 = *(const float4*)(scb + ko);
            float4 sc1 = *(const float4*)(scb + ko + 4);
            int4 mk0 = *(const int4*)(mkb + ko);
            int4 mk1 = *(const int4*)(mkb + ko + 4);
            float scv[8] = {sc0.x, sc0.y, sc0.z, sc0.w, sc1.x, sc1.y, sc1.z, sc1.w};
            int   mkv[8] = {mk0.x, mk0.y, mk0.z, mk0.w, mk1.x, mk1.y, mk1.z, mk1.w};
            #pragma unroll
            for (int r = 0; r < 8; r++) {
                float x = s[t][r] * 0.0625f + 0.5f * scv[r];  // (qk/8 + score)/2
                x = (mkv[r] == 0) ? -1.0e9f : x;
                xv[t][r] = x;
                tmax = fmaxf(tmax, x);
            }
        }
        tmax = fmaxf(tmax, __shfl_xor(tmax, 16, 32));
        float newm = fmaxf(mrun, tmax);
        float alpha = __expf(mrun - newm);
        float p0[8], p1[8], psum = 0.0f;
        #pragma unroll
        for (int r = 0; r < 8; r++) {
            p0[r] = __expf(xv[0][r] - newm);
            p1[r] = __expf(xv[1][r] - newm);
            psum += p0[r] + p1[r];
        }
        psum += __shfl_xor(psum, 16, 32);
        lrun = lrun * alpha + psum;
        mrun = newm;
        #pragma unroll
        for (int i = 0; i < 4; i++)
            #pragma unroll
            for (int r = 0; r < 8; r++) { oS[i][r] *= alpha; oO[i][r] *= alpha; }

        // --- pack P into B-operand fragment (K-dim = 32 keys) ---
        Frag16 pf;
        #pragma unroll
        for (int j = 0; j < 4; j++) {
            unsigned a  = pk_bf16(p0[2 * j], p0[2 * j + 1]);
            unsigned bb = pk_bf16(p1[2 * j], p1[2 * j + 1]);
            unsigned asw = __shfl_xor(a, 16, 32);
            unsigned bsw = __shfl_xor(bb, 16, 32);
            pf.us[j]     = a;
            pf.us[4 + j] = bb;
        }
        // --- O^T += V^T @ P^T  (both streams) ---
        #pragma unroll
        for (int dvt = 0; dvt < 4; dvt++) {
            size_t vrow = ((size_t)(b * 16 + h) * 64 + dvt * 16 + l15) * 2048 + k0 + 8 * lh;
            Frag16 vf;
            vf.u[0] = *(const uint4*)(VtS + vrow);
            vf.u[1] = *(const uint4*)(VtS + vrow + 16);
            oS[dvt] = wmma_bf16(vf.v, pf.v, oS[dvt]);
            vf.u[0] = *(const uint4*)(VtO + vrow);
            vf.u[1] = *(const uint4*)(VtO + vrow + 16);
            oO[dvt] = wmma_bf16(vf.v, pf.v, oO[dvt]);
        }
    }

    float invl = 1.0f / lrun;
    int row = b * 2048 + q0 + l15;
    #pragma unroll
    for (int dvt = 0; dvt < 4; dvt++) {
        Pack8 ps, po;
        #pragma unroll
        for (int r = 0; r < 8; r++) {
            ps.h[r] = (__bf16)(oS[dvt][r] * invl);
            po.h[r] = (__bf16)(oO[dvt][r] * invl);
        }
        size_t off = (size_t)row * 1024 + h * 64 + dvt * 16 + 8 * lh;
        *(uint4*)(Os + off) = ps.u;
        *(uint4*)(Oo + off) = po.u;
    }
}

// ---------------------------------------------------------------------------
// out = LayerNorm(O @ w_fc + residual) * gamma + beta     (fp32 out)
// Block: 8 waves, 16 rows x 1024 cols; wave handles 128 cols.
// ---------------------------------------------------------------------------
__global__ __launch_bounds__(256) void fc_ln_kernel(const __bf16* __restrict__ O,
                                                    const __bf16* __restrict__ WfcT,
                                                    const float* __restrict__ residual,
                                                    const float* __restrict__ gamma,
                                                    const float* __restrict__ beta,
                                                    float* __restrict__ out) {
    __shared__ float s1[8][16], s2[8][16], tot1[16], tot2[16];
    const int wave = threadIdx.x >> 5;
    const int lane = threadIdx.x & 31;
    const int l15 = lane & 15, lh = lane >> 4;
    const int m0 = blockIdx.x * 16;
    const int n0 = wave * 128;

    const v8f zero = {0.f, 0.f, 0.f, 0.f, 0.f, 0.f, 0.f, 0.f};
    v8f acc[8];
    #pragma unroll
    for (int i = 0; i < 8; i++) acc[i] = zero;

    for (int k0 = 0; k0 < 1024; k0 += 32) {
        Frag16 a;
        const __bf16* ap = O + (size_t)(m0 + l15) * 1024 + k0 + 8 * lh;
        a.u[0] = *(const uint4*)ap;
        a.u[1] = *(const uint4*)(ap + 16);
        #pragma unroll
        for (int ni = 0; ni < 8; ni++) {
            Frag16 bf;
            const __bf16* bp = WfcT + (size_t)(n0 + ni * 16 + l15) * 1024 + k0 + 8 * lh;
            bf.u[0] = *(const uint4*)bp;
            bf.u[1] = *(const uint4*)(bp + 16);
            acc[ni] = wmma_bf16(a.v, bf.v, acc[ni]);
        }
    }

    // x = acc + residual; per-row partial stats
    float rs[8], rs2[8];
    #pragma unroll
    for (int r = 0; r < 8; r++) {
        int row = m0 + r + 8 * lh;
        float sum = 0.f, sq = 0.f;
        #pragma unroll
        for (int ni = 0; ni < 8; ni++) {
            float x = acc[ni][r] + residual[(size_t)row * 1024 + n0 + ni * 16 + l15];
            acc[ni][r] = x;
            sum += x; sq += x * x;
        }
        rs[r] = sum; rs2[r] = sq;
    }
    #pragma unroll
    for (int r = 0; r < 8; r++)
        #pragma unroll
        for (int off = 1; off < 16; off <<= 1) {
            rs[r]  += __shfl_xor(rs[r],  off, 32);
            rs2[r] += __shfl_xor(rs2[r], off, 32);
        }
    if (l15 == 0) {
        #pragma unroll
        for (int r = 0; r < 8; r++) { s1[wave][8 * lh + r] = rs[r]; s2[wave][8 * lh + r] = rs2[r]; }
    }
    __syncthreads();
    if (threadIdx.x < 16) {
        float t = 0.f, t2 = 0.f;
        #pragma unroll
        for (int w = 0; w < 8; w++) { t += s1[w][threadIdx.x]; t2 += s2[w][threadIdx.x]; }
        tot1[threadIdx.x] = t; tot2[threadIdx.x] = t2;
    }
    __syncthreads();

    const float invD = 1.0f / 1024.0f;
    #pragma unroll
    for (int ni = 0; ni < 8; ni++) {
        int col = n0 + ni * 16 + l15;
        float g = gamma[col], bt = beta[col];
        #pragma unroll
        for (int r = 0; r < 8; r++) {
            int ri = r + 8 * lh;
            float mu = tot1[ri] * invD;
            float var = tot2[ri] * invD - mu * mu;
            float rstd = rsqrtf(var + 1e-6f);
            out[(size_t)(m0 + ri) * 1024 + col] = (acc[ni][r] - mu) * rstd * g + bt;
        }
    }
}

// ---------------------------------------------------------------------------
extern "C" void kernel_launch(void* const* d_in, const int* in_sizes, int n_in,
                              void* d_out, int out_size, void* d_ws, size_t ws_size,
                              hipStream_t stream) {
    const float* PE_states     = (const float*)d_in[0];
    const float* global_state  = (const float*)d_in[1];
    const float* PE_statements = (const float*)d_in[2];
    const float* PE_operators  = (const float*)d_in[3];
    const float* scores        = (const float*)d_in[4];
    const int*   mask          = (const int*)d_in[5];
    const float* w_q  = (const float*)d_in[6];
    const float* w_k  = (const float*)d_in[7];
    const float* w_v  = (const float*)d_in[8];
    const float* w_fc = (const float*)d_in[9];
    const float* ln_g = (const float*)d_in[10];
    const float* ln_b = (const float*)d_in[11];

    char* ws = (char*)d_ws;
    const size_t MB = 1u << 20;
    __bf16* A0  = (__bf16*)(ws + 0 * MB);   // global_state bf16  -> later Os
    __bf16* A1  = (__bf16*)(ws + 8 * MB);   // PE_states bf16     -> later Oo
    __bf16* A2  = (__bf16*)(ws + 16 * MB);  // PE_statements bf16
    __bf16* A3  = (__bf16*)(ws + 24 * MB);  // PE_operators bf16
    __bf16* WQ  = (__bf16*)(ws + 32 * MB);
    __bf16* WK  = (__bf16*)(ws + 34 * MB);
    __bf16* WV  = (__bf16*)(ws + 36 * MB);
    __bf16* WFC = (__bf16*)(ws + 38 * MB);
    __bf16* QB  = (__bf16*)(ws + 40 * MB);
    __bf16* KB  = (__bf16*)(ws + 48 * MB);
    __bf16* VTS = (__bf16*)(ws + 56 * MB);
    __bf16* VTO = (__bf16*)(ws + 64 * MB);
    __bf16* OS  = A0;
    __bf16* OO  = A1;

    const int n4 = (2 * 2048 * 1024) / 4;  // 1,048,576 float4 groups
    cast_bf16_kernel<<<4096, 256, 0, stream>>>(global_state,  A0, n4);
    cast_bf16_kernel<<<4096, 256, 0, stream>>>(PE_states,     A1, n4);
    cast_bf16_kernel<<<4096, 256, 0, stream>>>(PE_statements, A2, n4);
    cast_bf16_kernel<<<4096, 256, 0, stream>>>(PE_operators,  A3, n4);

    dim3 tg(32, 32), tb(32, 8);
    transpose_cast_kernel<<<tg, tb, 0, stream>>>(w_q,  WQ,  1024, 1024);
    transpose_cast_kernel<<<tg, tb, 0, stream>>>(w_k,  WK,  1024, 1024);
    transpose_cast_kernel<<<tg, tb, 0, stream>>>(w_v,  WV,  1024, 1024);
    transpose_cast_kernel<<<tg, tb, 0, stream>>>(w_fc, WFC, 1024, 1024);

    dim3 gg(4, 64);  // N/256 x M/64
    gemm_bf16_kernel<<<gg, 256, 0, stream>>>(A0, WQ, QB,  4096, 1024, 1024, 0, 2048);
    gemm_bf16_kernel<<<gg, 256, 0, stream>>>(A1, WK, KB,  4096, 1024, 1024, 0, 2048);
    gemm_bf16_kernel<<<gg, 256, 0, stream>>>(A2, WV, VTS, 4096, 1024, 1024, 1, 2048);
    gemm_bf16_kernel<<<gg, 256, 0, stream>>>(A3, WV, VTO, 4096, 1024, 1024, 1, 2048);

    attn_kernel<<<1024, 128, 0, stream>>>(QB, KB, VTS, VTO, scores, mask, OS, OO);

    float* out0 = (float*)d_out;
    float* out1 = out0 + (size_t)2 * 2048 * 1024;
    fc_ln_kernel<<<256, 256, 0, stream>>>(OS, WFC, global_state, ln_g, ln_b, out0);
    fc_ln_kernel<<<256, 256, 0, stream>>>(OO, WFC, global_state, ln_g, ln_b, out1);
}
